// Head_40896678592848
// MI455X (gfx1250) — matrix operand order repaired and checked by reading, not compiled, not run on hardware
//
#include <hip/hip_runtime.h>
#include <hip/hip_bf16.h>

// ---------------------------------------------------------------------------
// Causal attention head, bf16 WMMA flash-attention style, gfx1250 (CDNA5).
// B=4, T=4096, E=1024, A=64.  wave32, V_WMMA_F32_16X16X32_BF16 throughout.
// ---------------------------------------------------------------------------

#define B_ 4
#define T_ 4096
#define E_ 1024
#define A_ 64
#define NEG_BIG (-1.0e30f)

typedef __attribute__((ext_vector_type(16))) __bf16 v16bf;
typedef __attribute__((ext_vector_type(8)))  float  v8f;

union Frag {
  unsigned u[8];
  uint4    q[2];
  v16bf    v;
};

// round-to-nearest (ties away from zero): 3 VALU per pair instead of ~7 for RNE
__device__ __forceinline__ unsigned short f32_to_bf16_rn(float f) {
  return (unsigned short)((__float_as_uint(f) + 0x8000u) >> 16);
}

__device__ __forceinline__ unsigned pack2_bf16(float lo, float hi) {
  unsigned a = __float_as_uint(lo) + 0x8000u;
  unsigned b = __float_as_uint(hi) + 0x8000u;
  return (a >> 16) | (b & 0xFFFF0000u);
}

__device__ __forceinline__ v8f wmma_bf16(const Frag& a, const Frag& b, v8f c) {
  // (neg_a, A, neg_b, B, c_mod, C, reuse_a, reuse_b)
  return __builtin_amdgcn_wmma_f32_16x16x32_bf16(false, a.v, false, b.v,
                                                 (short)0, c, false, false);
}

// ---------------------------------------------------------------------------
// Kernel 0: convert fp32 weights [E][A] -> bf16 transposed [A][E], all three.
// ---------------------------------------------------------------------------
__global__ __launch_bounds__(256)
void prep_weights_kernel(const float* __restrict__ wq,
                         const float* __restrict__ wk,
                         const float* __restrict__ wv,
                         unsigned short* __restrict__ wt) {
  int idx   = blockIdx.x * blockDim.x + threadIdx.x;  // [0, 3*A_*E_)
  int which = idx / (A_ * E_);
  int rem   = idx % (A_ * E_);
  int a = rem / E_;
  int e = rem % E_;
  const float* w = (which == 0) ? wq : (which == 1) ? wk : wv;
  wt[idx] = f32_to_bf16_rn(w[e * A_ + a]);
}

// ---------------------------------------------------------------------------
// Kernel 1: projections  X[B*T][E] @ W[E][A] -> bf16.
//   which = blockIdx.y: 0->q [T][A], 1->k [T][A], 2->v transposed [A][T].
//   Block: 128 threads (4 waves). Wave w: rows [64*bx + 16w, +16), cols 0..63.
//   Software-pipelined A loads; B fragment loads batched ahead of the WMMAs.
// ---------------------------------------------------------------------------
__global__ __launch_bounds__(128)
void proj_kernel(const float* __restrict__ q_in,
                 const float* __restrict__ k_in,
                 const float* __restrict__ v_in,
                 const unsigned short* __restrict__ wt_all,
                 unsigned short* __restrict__ qp,
                 unsigned short* __restrict__ kp,
                 unsigned short* __restrict__ vt) {
  const int lane  = threadIdx.x & 31;
  const int wave  = threadIdx.x >> 5;
  const int which = blockIdx.y;
  const float* x = (which == 0) ? q_in : (which == 1) ? k_in : v_in;
  const unsigned short* wt = wt_all + which * (A_ * E_);

  const long row0 = (long)blockIdx.x * 64 + wave * 16;   // global row in [0, B*T)
  const int  mcol = lane & 15;                            // M-row for A, N-col for B/C
  const int  half = lane >> 4;

  v8f acc[4] = {};
  const float* arow = x + (row0 + mcol) * E_ + (half << 3);

  // prologue: first A chunk
  float4 f0 = *(const float4*)(arow);
  float4 f1 = *(const float4*)(arow + 4);
  float4 f2 = *(const float4*)(arow + 16);
  float4 f3 = *(const float4*)(arow + 20);

  for (int k0 = 0; k0 < E_; k0 += 32) {
    // A fragment (16x32 bf16): lanes 0-15 K=0..7,16..23 ; lanes 16-31 K=8..15,24..31
    Frag af;
    af.u[0] = pack2_bf16(f0.x, f0.y); af.u[1] = pack2_bf16(f0.z, f0.w);
    af.u[2] = pack2_bf16(f1.x, f1.y); af.u[3] = pack2_bf16(f1.z, f1.w);
    af.u[4] = pack2_bf16(f2.x, f2.y); af.u[5] = pack2_bf16(f2.z, f2.w);
    af.u[6] = pack2_bf16(f3.x, f3.y); af.u[7] = pack2_bf16(f3.z, f3.w);

    // pipeline: issue next A-chunk loads before this iteration's WMMAs
    if (k0 + 32 < E_) {
      f0 = *(const float4*)(arow + k0 + 32);
      f1 = *(const float4*)(arow + k0 + 36);
      f2 = *(const float4*)(arow + k0 + 48);
      f3 = *(const float4*)(arow + k0 + 52);
    }

    // batch all 4 B-fragment loads, then 4 back-to-back WMMAs
    Frag bf[4];
#pragma unroll
    for (int j = 0; j < 4; ++j) {
      const unsigned short* brow = wt + (j * 16 + mcol) * E_ + k0 + (half << 3);
      bf[j].q[0] = *(const uint4*)(brow);
      bf[j].q[1] = *(const uint4*)(brow + 16);
    }
#pragma unroll
    for (int j = 0; j < 4; ++j) acc[j] = wmma_bf16(af, bf[j], acc[j]);
  }

  // C layout: VGPR r, lanes 0-15: M=r ; lanes 16-31: M=r+8 ; N = lane&15
  const int mhalf = half << 3;
  if (which == 2) {                       // v transposed: vt[b][a][t]
    const long bb = row0 >> 12;           // 64-row tile never straddles a batch
    const long t0 = (row0 & (T_ - 1)) + mhalf;
#pragma unroll
    for (int j = 0; j < 4; ++j) {
      const long base = (bb * A_ + j * 16 + mcol) * T_ + t0;
#pragma unroll
      for (int r = 0; r < 8; ++r)
        vt[base + r] = f32_to_bf16_rn(acc[j][r]);
    }
  } else {                                // q/k row-major [T][A]
    unsigned short* dst = which ? kp : qp;
#pragma unroll
    for (int j = 0; j < 4; ++j) {
#pragma unroll
      for (int r = 0; r < 8; ++r)
        dst[(row0 + r + mhalf) * A_ + j * 16 + mcol] = f32_to_bf16_rn(acc[j][r]);
    }
  }
}

// ---------------------------------------------------------------------------
// Kernel 2: streaming causal attention.
//   Grid: (T_/64, B_). Block: 128 threads (4 waves). Wave owns 16 query rows.
//   Loops over 64-wide key blocks up to causal diagonal; online softmax.
// ---------------------------------------------------------------------------
__global__ __launch_bounds__(128)
void attn_kernel(const unsigned short* __restrict__ qp,
                 const unsigned short* __restrict__ kp,
                 const unsigned short* __restrict__ vt,
                 float* __restrict__ out) {
  __shared__ __align__(16) unsigned short p_lds[4][16][64];   // 8 KB

  const int lane = threadIdx.x & 31;
  const int wave = threadIdx.x >> 5;
  const int b    = blockIdx.y;
  const int qb   = blockIdx.x * 64 + wave * 16;  // query row base within T

  const unsigned short* q_base = qp + (long)b * T_ * A_;
  const unsigned short* k_base = kp + (long)b * T_ * A_;
  const unsigned short* v_base = vt + (long)b * A_ * T_;

  const int mcol  = lane & 15;
  const int half  = lane >> 4;
  const int mhalf = half << 3;

  // Q fragments for this wave's 16 rows, K(feature)=64 -> two 16x32 fragments.
  Frag qf[2];
  {
    const unsigned short* qrow = q_base + (qb + mcol) * A_ + (half << 3);
#pragma unroll
    for (int c = 0; c < 2; ++c) {
      qf[c].q[0] = *(const uint4*)(qrow + 32 * c);
      qf[c].q[1] = *(const uint4*)(qrow + 32 * c + 16);
    }
  }

  v8f oacc[4] = {};
  float mrow[8], lrow[8];
#pragma unroll
  for (int r = 0; r < 8; ++r) { mrow[r] = NEG_BIG; lrow[r] = 0.f; }

  const int nkb = (qb + 15) / 64 + 1;   // key blocks intersecting causal region
  for (int kb = 0; kb < nkb; ++kb) {
    const int kbase = kb * 64;

    // prefetch next key block (K rows + V^T rows) into the cache hierarchy
    if (kb + 1 < nkb) {
      __builtin_prefetch(k_base + (kbase + 64 + 2 * lane) * A_, 0, 1);
      __builtin_prefetch(v_base + (2 * lane) * T_ + kbase + 64, 0, 1);
    }

    // ---- scores: S = Q @ K^T  (16x64 per wave) ----
    v8f sacc[4] = {};
#pragma unroll
    for (int c = 0; c < 2; ++c) {
      Frag kf[4];  // B operand: column n = key (kbase+16j+mcol), K = features
#pragma unroll
      for (int j = 0; j < 4; ++j) {
        const unsigned short* krow =
            k_base + (kbase + j * 16 + mcol) * A_ + 32 * c + (half << 3);
        kf[j].q[0] = *(const uint4*)(krow);
        kf[j].q[1] = *(const uint4*)(krow + 16);
      }
#pragma unroll
      for (int j = 0; j < 4; ++j) sacc[j] = wmma_bf16(qf[c], kf[j], sacc[j]);
    }

    // ---- scale + causal mask + per-row block max ----
    float pm[8];
#pragma unroll
    for (int r = 0; r < 8; ++r) {
      const int qg = qb + r + mhalf;
      float rmax = NEG_BIG;
#pragma unroll
      for (int j = 0; j < 4; ++j) {
        float s = sacc[j][r] * 0.125f;           // 1/sqrt(64)
        int   n = kbase + j * 16 + mcol;
        if (n > qg) s = NEG_BIG;
        sacc[j][r] = s;
        rmax = fmaxf(rmax, s);
      }
      pm[r] = rmax;
    }
#pragma unroll
    for (int r = 0; r < 8; ++r) {                // reduce across 16-lane group
      float v = pm[r];
      v = fmaxf(v, __shfl_xor(v, 1, 32));
      v = fmaxf(v, __shfl_xor(v, 2, 32));
      v = fmaxf(v, __shfl_xor(v, 4, 32));
      v = fmaxf(v, __shfl_xor(v, 8, 32));
      pm[r] = v;
    }

    // ---- online softmax rescale ----
#pragma unroll
    for (int r = 0; r < 8; ++r) {
      float mn = fmaxf(mrow[r], pm[r]);
      float sc = __expf(mrow[r] - mn);
      mrow[r] = mn;
      lrow[r] *= sc;
#pragma unroll
      for (int j = 0; j < 4; ++j) oacc[j][r] *= sc;
    }

    // ---- P = exp(S - m), row sums, stage bf16 P in LDS (C -> A layout) ----
    float psum[8];
#pragma unroll
    for (int r = 0; r < 8; ++r) psum[r] = 0.f;
#pragma unroll
    for (int j = 0; j < 4; ++j) {
#pragma unroll
      for (int r = 0; r < 8; ++r) {
        float p = __expf(sacc[j][r] - mrow[r]);
        psum[r] += p;
        p_lds[wave][r + mhalf][j * 16 + mcol] = f32_to_bf16_rn(p);
      }
    }
#pragma unroll
    for (int r = 0; r < 8; ++r) {
      float v = psum[r];
      v += __shfl_xor(v, 1, 32);
      v += __shfl_xor(v, 2, 32);
      v += __shfl_xor(v, 4, 32);
      v += __shfl_xor(v, 8, 32);
      lrow[r] += v;
    }

    // per-wave LDS region; DS ops are in-order, force completion before reads
    asm volatile("s_wait_dscnt 0x0" ::: "memory");

    // ---- read P as A fragments, O += P @ V ----
    Frag pf[2];
    const unsigned short* prow = &p_lds[wave][mcol][half << 3];
#pragma unroll
    for (int c = 0; c < 2; ++c) {
      pf[c].q[0] = *(const uint4*)(prow + 32 * c);
      pf[c].q[1] = *(const uint4*)(prow + 32 * c + 16);
    }
#pragma unroll
    for (int c = 0; c < 2; ++c) {
      Frag vf[4];  // B operand: column n = feature (16j+mcol), K = keys
#pragma unroll
      for (int j = 0; j < 4; ++j) {
        const unsigned short* vrow =
            v_base + (j * 16 + mcol) * T_ + kbase + 32 * c + (half << 3);
        vf[j].q[0] = *(const uint4*)(vrow);
        vf[j].q[1] = *(const uint4*)(vrow + 16);
      }
#pragma unroll
      for (int j = 0; j < 4; ++j) oacc[j] = wmma_bf16(pf[c], vf[j], oacc[j]);
    }
    asm volatile("s_wait_dscnt 0x0" ::: "memory");  // reads done before rewrite
  }

  // ---- finalize: out[b][qb+m][f] = O * (1/l) ----
  float linv[8];
#pragma unroll
  for (int r = 0; r < 8; ++r) linv[r] = 1.0f / lrow[r];
#pragma unroll
  for (int j = 0; j < 4; ++j) {
#pragma unroll
    for (int r = 0; r < 8; ++r) {
      int m = r + mhalf;
      out[((long)b * T_ + qb + m) * A_ + j * 16 + mcol] = oacc[j][r] * linv[r];
    }
  }
}

// ---------------------------------------------------------------------------
// Host launcher
// ---------------------------------------------------------------------------
extern "C" void kernel_launch(void* const* d_in, const int* in_sizes, int n_in,
                              void* d_out, int out_size, void* d_ws, size_t ws_size,
                              hipStream_t stream) {
  (void)in_sizes; (void)n_in; (void)out_size; (void)ws_size;
  const float* query = (const float*)d_in[0];
  const float* key   = (const float*)d_in[1];
  const float* value = (const float*)d_in[2];
  const float* wq    = (const float*)d_in[3];
  const float* wk    = (const float*)d_in[4];
  const float* wv    = (const float*)d_in[5];
  float* out = (float*)d_out;

  // workspace layout (bf16/ushort): Wt[3][A][E] | q[B*T][A] | k[B*T][A] | vT[B][A][T]
  unsigned short* wt = (unsigned short*)d_ws;
  unsigned short* qp = wt + 3 * A_ * E_;
  unsigned short* kp = qp + (long)B_ * T_ * A_;
  unsigned short* vt = kp + (long)B_ * T_ * A_;

  prep_weights_kernel<<<(3 * A_ * E_) / 256, 256, 0, stream>>>(wq, wk, wv, wt);
  proj_kernel<<<dim3((B_ * T_) / 64, 3), 128, 0, stream>>>(query, key, value,
                                                           wt, qp, kp, vt);
  attn_kernel<<<dim3(T_ / 64, B_), 128, 0, stream>>>(qp, kp, vt, out);
}
